// VectorQuantizer_64398739636824
// MI455X (gfx1250) — matrix operand order repaired and checked by reading, not compiled, not run on hardware
//
#include <hip/hip_runtime.h>
#include <hip/hip_bf16.h>

// VQ-VAE vector quantizer for MI455X (gfx1250, wave32, WMMA).
// dists = ||x||^2 + ||e||^2 - 2 x.e  -> argmin_k (||e_k||^2 - 2 x.e_k)
// Output = exact f32 codebook rows (straight-through value == quantized),
// so f16 WMMA only decides the argmin; values stay exact f32.
//
// Roofline: 67 MB traffic @ 23.3 TB/s ~ 2.9 us; 8.6 GFLOP via f16 WMMA << that.
// E16 staged to LDS with GLOBAL_LOAD_ASYNC_TO_LDS_B128 (ASYNCcnt path).

typedef __attribute__((ext_vector_type(16))) _Float16 v16h;
typedef __attribute__((ext_vector_type(8)))  _Float16 v8h;
typedef __attribute__((ext_vector_type(8)))  float    v8f;

#define VQ_D   128
#define VQ_K   512
#define VQ_HW  4096   // H*W = 64*64

// ---------------- prep: E(f32) -> E(f16) + ||e||^2 ----------------
__global__ __launch_bounds__(32)
void vq_prep_kernel(const float* __restrict__ E,
                    _Float16* __restrict__ e16,
                    float* __restrict__ enorm) {
    const int k = blockIdx.x;      // 0..511
    const int t = threadIdx.x;     // 0..31, one wave per code
    const float4 v = ((const float4*)(E + (size_t)k * VQ_D))[t];
    _Float16* dst = e16 + (size_t)k * VQ_D + t * 4;
    dst[0] = (_Float16)v.x; dst[1] = (_Float16)v.y;
    dst[2] = (_Float16)v.z; dst[3] = (_Float16)v.w;
    float s = v.x * v.x + v.y * v.y + v.z * v.z + v.w * v.w;
    #pragma unroll
    for (int off = 16; off > 0; off >>= 1) s += __shfl_xor(s, off, 32);
    if (t == 0) enorm[k] = s;
}

// ---------------- main: distances via WMMA + argmin + gather ----------------
__global__ __launch_bounds__(256)
void vq_main_kernel(const float* __restrict__ X,        // [B, D, H, W] f32
                    const float* __restrict__ Ew,       // [K, D] f32
                    const _Float16* __restrict__ E16,   // [K, D] f16 (ws)
                    const float* __restrict__ enormG,   // [K] f32 (ws)
                    float* __restrict__ out) {          // [B, D, H, W] f32
    // 128 codes staged per chunk; row stride 136 halves (272B = 17*16B) keeps
    // ds_load_b128 16B-aligned with <=2-way bank conflicts.
    __shared__ __align__(16) _Float16 eLds[128 * 136];
    __shared__ float sEn[VQ_K];
    __shared__ int   sIdx[128];

    const int tid  = threadIdx.x;
    const int lane = tid & 31;
    const int wave = tid >> 5;       // 8 waves, 16 rows each
    const int half = lane >> 4;      // 0 | 1
    const int l16  = lane & 15;

    const int nbase = blockIdx.x * 128;      // 128 consecutive flat positions
    const int b  = nbase >> 12;              // never crosses a batch boundary
    const int hw = nbase & (VQ_HW - 1);

    // stage ||e||^2 (2 per thread)
    sEn[tid]       = enormG[tid];
    sEn[tid + 256] = enormG[tid + 256];

    // Generic->LDS byte offset: low 32 bits of the flat address (ISA 10.2).
    const unsigned ldsBase = (unsigned)(uintptr_t)&eLds[0];

    // ---- A tile: 16 rows x 128 d per wave, ISA 16-bit A 16x32 layout ----
    // lane holds row = l16; kbase = 8*half; v16h[j]=K kbase+j, v16h[8+j]=K kbase+16+j
    const float* xin = X + (size_t)b * (VQ_D * VQ_HW) + hw + wave * 16 + l16;
    const int kbase = half * 8;
    v16h a[4];
    #pragma unroll
    for (int kt = 0; kt < 4; ++kt) {
        #pragma unroll
        for (int j = 0; j < 8; ++j) {
            const int d0 = kt * 32 + kbase + j;
            a[kt][j]     = (_Float16)xin[(size_t)d0 * VQ_HW];
            a[kt][j + 8] = (_Float16)xin[(size_t)(d0 + 16) * VQ_HW];
        }
    }

    float bestv[8];
    int   besti[8];
    #pragma unroll
    for (int r = 0; r < 8; ++r) { bestv[r] = 3.0e38f; besti[r] = 0; }

    for (int c = 0; c < 4; ++c) {
        __syncthreads();  // previous chunk fully consumed (also covers sEn)
        // stage 128 codes (32KB) -> LDS via async global->LDS DMA (ASYNCcnt).
        // Each lane moves 16B per issue; 8 issues * 256 threads = 32KB.
        {
            #pragma unroll
            for (int i = 0; i < 8; ++i) {
                const int idx = tid + i * 256;      // 0..2047 16B chunks
                const int row = idx >> 4;
                const int col = idx & 15;
                const unsigned dstOff =
                    ldsBase + (unsigned)(row * 136 + col * 8) * 2u;
                const unsigned long long gaddr = (unsigned long long)(uintptr_t)
                    (E16 + (size_t)c * 128 * VQ_D + row * VQ_D + col * 8);
                asm volatile("global_load_async_to_lds_b128 %0, %1, off"
                             :: "v"(dstOff), "v"(gaddr) : "memory");
            }
            asm volatile("s_wait_asynccnt 0x0" ::: "memory");
        }
        __syncthreads();

        #pragma unroll
        for (int jt = 0; jt < 8; ++jt) {
            const int codeRow = jt * 16 + l16;       // B layout: lane = column
            const int code    = c * 128 + codeRow;
            const float en    = sEn[code];
            v8f acc = {};
            #pragma unroll
            for (int kt = 0; kt < 4; ++kt) {
                // B 32x16 f16 layout: h[0] = K 8*half..+7, h[1] = K 16+8*half..+7
                union { v16h v; v8h h[2]; } bb;
                const _Float16* bp = &eLds[codeRow * 136 + kt * 32 + half * 8];
                bb.h[0] = *(const v8h*)bp;
                bb.h[1] = *(const v8h*)(bp + 16);
                acc = __builtin_amdgcn_wmma_f32_16x16x32_f16(
                        false, a[kt], false, bb.v, (short)0, acc, false, false);
            }
            // C layout: reg r, lane -> row M = r + 8*half, col N = l16 (=code)
            #pragma unroll
            for (int r = 0; r < 8; ++r) {
                const float s = en - 2.0f * acc[r];
                if (s < bestv[r]) { bestv[r] = s; besti[r] = code; }  // ascending codes => first-min
            }
        }
    }

    // argmin across the 16 lanes of each half (rows r / r+8); tie -> lower index
    #pragma unroll
    for (int r = 0; r < 8; ++r) {
        #pragma unroll
        for (int off = 1; off < 16; off <<= 1) {
            const float ov = __shfl_xor(bestv[r], off, 32);
            const int   oi = __shfl_xor(besti[r], off, 32);
            if (ov < bestv[r] || (ov == bestv[r] && oi < besti[r])) {
                bestv[r] = ov; besti[r] = oi;
            }
        }
    }
    if (l16 == 0) {
        #pragma unroll
        for (int r = 0; r < 8; ++r) sIdx[wave * 16 + r + half * 8] = besti[r];
    }
    __syncthreads();

    // gather: out[b, d, hw+nl] = Ew[idx[nl], d]; float4 writes, fully coalesced
    float* op = out + (size_t)b * (VQ_D * VQ_HW) + hw;
    for (int i = tid; i < 128 * 32; i += 256) {
        const int d  = i >> 5;
        const int nl = (i & 31) * 4;
        float4 v;
        v.x = Ew[(size_t)sIdx[nl + 0] * VQ_D + d];
        v.y = Ew[(size_t)sIdx[nl + 1] * VQ_D + d];
        v.z = Ew[(size_t)sIdx[nl + 2] * VQ_D + d];
        v.w = Ew[(size_t)sIdx[nl + 3] * VQ_D + d];
        *(float4*)(op + (size_t)d * VQ_HW + nl) = v;
    }
}

extern "C" void kernel_launch(void* const* d_in, const int* in_sizes, int n_in,
                              void* d_out, int out_size, void* d_ws, size_t ws_size,
                              hipStream_t stream) {
    (void)in_sizes; (void)n_in; (void)out_size; (void)ws_size;
    const float* X  = (const float*)d_in[0];   // inputs [16,128,64,64] f32
    const float* Ew = (const float*)d_in[1];   // embedding_weight [512,128] f32
    float* out = (float*)d_out;

    _Float16* e16 = (_Float16*)d_ws;                                  // 128 KB
    float* enorm  = (float*)((char*)d_ws + (size_t)VQ_K * VQ_D * 2);  // 2 KB

    vq_prep_kernel<<<VQ_K, 32, 0, stream>>>(Ew, e16, enorm);
    // 65536 rows / 128 per block = 512 blocks
    vq_main_kernel<<<512, 256, 0, stream>>>(X, Ew, e16, enorm, out);
}
